// Timm_31722628448632
// MI455X (gfx1250) — matrix operand order repaired and checked by reading, not compiled, not run on hardware
//
#include <hip/hip_runtime.h>

typedef __attribute__((ext_vector_type(16))) __bf16 v16bf;
typedef __attribute__((ext_vector_type(8)))  float  v8f;

#define NA_TOTAL 49104
#define NUM_CLASSES 80
#define KTOP 300
#define CLS_TH 0.05f
#define NMS_TH 0.5f
#define KDIM 2304             // 256 * 9
#define CI_CHUNK 64
#define CHUNK_K (CI_CHUNK*9)  // 576
#define TILE_N 32             // two 16-wide WMMA N tiles per block

union V16 { v16bf v; uint4 q[2]; };

// ---------------- fp32 -> bf16 conversion ----------------
__global__ __launch_bounds__(256) void f32_to_bf16(const float* __restrict__ src,
                                                   __bf16* __restrict__ dst, int n) {
  int i = blockIdx.x * 256 + threadIdx.x;
  if (i < n) dst[i] = (__bf16)src[i];
}

// ---------------- implicit-GEMM conv3x3 via WMMA ----------------
// actIn: bf16 [256][HW], w: bf16 [outC][2304], bias: f32 [outC]
// mode 0: hidden layer, relu -> bf16 actOut [outC][HW]
// mode 1: regression output (outC=36=9*4) -> headOut[(aoff+pix*9+a)*4+j]
// mode 2: classification output (outC=720=9*80), sigmoid -> headOut[(aoff+pix*9+a)*80+c]
__global__ __launch_bounds__(256) void conv3x3_wmma(
    const __bf16* __restrict__ actIn, const __bf16* __restrict__ w,
    const float* __restrict__ bias, int H, int outC, int mode,
    __bf16* __restrict__ actOut, float* __restrict__ headOut, int aoff)
{
  __shared__ __attribute__((aligned(16))) __bf16 patch[TILE_N][CHUNK_K]; // 36 KB, [pixel][k]
  const int HW   = H * H;
  const int tid  = threadIdx.x;
  const int lane = tid & 31;
  const int wv   = tid >> 5;
  const int pix0 = blockIdx.x * TILE_N;
  const int ot   = blockIdx.y * 8 + wv;          // 16-wide out-channel tile index
  const bool active = (ot * 16) < outC;

  v8f acc[2];
  #pragma unroll
  for (int t = 0; t < 2; ++t)
    #pragma unroll
    for (int i = 0; i < 8; ++i) acc[t][i] = 0.0f;

  const int aoffk = (lane < 16) ? 0 : 8;         // A-matrix 16-bit layout
  const int boffk = (lane < 16) ? 0 : 8;         // B-matrix 16-bit layout
  const __bf16* wrowBase = w + (size_t)(ot * 16 + (lane & 15)) * KDIM;
  const __bf16* brow0 = &patch[lane & 15][0];
  const __bf16* brow1 = &patch[16 + (lane & 15)][0];

  for (int ph = 0; ph < 4; ++ph) {
    // ---- cooperative im2col staging: ci chunk [ph*64, ph*64+64) x 32 pixels ----
    for (int idx = tid; idx < CI_CHUNK * TILE_N; idx += 256) {
      int ciL = idx >> 5;          // 0..63
      int n   = idx & (TILE_N - 1);
      int pix = pix0 + n;
      bool pv = pix < HW;
      int y = pv ? (pix / H) : 0;
      int x = pv ? (pix % H) : 0;
      const __bf16* src = actIn + (size_t)(ph * CI_CHUNK + ciL) * HW;
      #pragma unroll
      for (int ky = 0; ky < 3; ++ky) {
        int yy = y + ky - 1;
        #pragma unroll
        for (int kx = 0; kx < 3; ++kx) {
          int xx = x + kx - 1;
          __bf16 val = (__bf16)0.0f;
          if (pv && yy >= 0 && yy < H && xx >= 0 && xx < H)
            val = src[yy * H + xx];
          patch[n][ciL * 9 + ky * 3 + kx] = val;
        }
      }
    }
    __syncthreads();

    if (active) {
      const __bf16* wrow = wrowBase + ph * CHUNK_K;
      __builtin_prefetch(wrow + CHUNK_K, 0, 1);  // global_prefetch_b8 for next phase
      #pragma unroll
      for (int ks = 0; ks < CHUNK_K / 32; ++ks) {  // 18 k-steps, 2 WMMA each
        int gk = ks * 32;
        V16 a, b0, b1;
        a.q[0] = *(const uint4*)(wrow + gk + aoffk);
        a.q[1] = *(const uint4*)(wrow + gk + 16 + aoffk);
        b0.q[0] = *(const uint4*)(brow0 + gk + boffk);
        b0.q[1] = *(const uint4*)(brow0 + gk + boffk + 16);
        acc[0] = __builtin_amdgcn_wmma_f32_16x16x32_bf16(false, a.v, false, b0.v,
                                                         (short)0, acc[0], false, false);
        b1.q[0] = *(const uint4*)(brow1 + gk + boffk);
        b1.q[1] = *(const uint4*)(brow1 + gk + boffk + 16);
        acc[1] = __builtin_amdgcn_wmma_f32_16x16x32_bf16(false, a.v, false, b1.v,
                                                         (short)0, acc[1], false, false);
      }
    }
    __syncthreads();
  }

  // ---- epilogue using C/D layout: VGPR r -> M = r + (lane<16 ? 0 : 8), N = lane%16 ----
  if (active) {
    int n = lane & 15;
    int mbase = ot * 16 + ((lane < 16) ? 0 : 8);
    #pragma unroll
    for (int t = 0; t < 2; ++t) {
      int pix = pix0 + t * 16 + n;
      if (pix < HW) {
        #pragma unroll
        for (int r = 0; r < 8; ++r) {
          int co = mbase + r;
          if (co < outC) {
            float v = acc[t][r] + bias[co];
            if (mode == 0) {
              v = fmaxf(v, 0.0f);
              actOut[(size_t)co * HW + pix] = (__bf16)v;
            } else if (mode == 1) {
              int a9 = co >> 2, j = co & 3;
              headOut[((size_t)(aoff + pix * 9 + a9)) * 4 + j] = v;
            } else {
              int a9 = co / 80, cc = co % 80;
              float s = 1.0f / (1.0f + __expf(-v));
              headOut[((size_t)(aoff + pix * 9 + a9)) * 80 + cc] = s;
            }
          }
        }
      }
    }
  }
}

// ---------------- anchor generation + box decode + clip ----------------
__global__ __launch_bounds__(256) void decode_boxes(const float* __restrict__ reg,
                                                    float* __restrict__ boxes)
{
  int A = blockIdx.x * 256 + threadIdx.x;
  if (A >= NA_TOTAL) return;
  int lvl, base;
  if      (A < 36864) { lvl = 0; base = 0; }
  else if (A < 46080) { lvl = 1; base = 36864; }
  else if (A < 48384) { lvl = 2; base = 46080; }
  else if (A < 48960) { lvl = 3; base = 48384; }
  else                { lvl = 4; base = 48960; }
  int rel = A - base;
  int pix = rel / 9, ai = rel % 9;
  int fs = 64 >> lvl;
  float stride = (float)(8 << lvl);
  float size   = (float)(32 << lvl);
  int y = pix / fs, x = pix % fs;
  float cxa = (x + 0.5f) * stride, cya = (y + 0.5f) * stride;
  int ri = ai / 3, si = ai % 3;
  float ratio = (ri == 0) ? 0.5f : (ri == 1 ? 1.0f : 2.0f);
  float scale = (si == 0) ? 1.0f : (si == 1 ? 1.2599210498948732f : 1.5874010519681994f);
  float w = size * scale / sqrtf(ratio);
  float h = w * ratio;
  float dx = reg[A * 4 + 0] * 0.1f, dy = reg[A * 4 + 1] * 0.1f;
  float dw = reg[A * 4 + 2] * 0.2f, dh = reg[A * 4 + 3] * 0.2f;
  float pcx = cxa + dx * w, pcy = cya + dy * h;
  float pw = __expf(dw) * w, ph = __expf(dh) * h;
  boxes[A * 4 + 0] = fminf(fmaxf(pcx - 0.5f * pw, 0.f), 512.f);
  boxes[A * 4 + 1] = fminf(fmaxf(pcy - 0.5f * ph, 0.f), 512.f);
  boxes[A * 4 + 2] = fminf(fmaxf(pcx + 0.5f * pw, 0.f), 512.f);
  boxes[A * 4 + 3] = fminf(fmaxf(pcy + 0.5f * ph, 0.f), 512.f);
}

// ---------------- mask + transpose: cls[A][80] -> clsT[80][A] ----------------
__global__ __launch_bounds__(256) void mask_transpose(const float* __restrict__ cls,
                                                      float* __restrict__ clsT, int A)
{
  int i = blockIdx.x * 256 + threadIdx.x;
  if (i < A * NUM_CLASSES) {
    int a = i / NUM_CLASSES, c = i % NUM_CLASSES;
    float s = cls[i];
    clsT[(size_t)c * A + a] = (s > CLS_TH) ? s : -1.0f;
  }
}

// ---------------- per-class top-300 (iterative block argmax) ----------------
__global__ __launch_bounds__(256) void topk_select(float* __restrict__ clsT,
                                                   const float* __restrict__ boxes,
                                                   float* __restrict__ topS,
                                                   float* __restrict__ topB,
                                                   int* __restrict__ topI)
{
  __shared__ float sV[256];
  __shared__ int   sI[256];
  int c = blockIdx.x, tid = threadIdx.x;
  float* sc = clsT + (size_t)c * NA_TOTAL;
  for (int k = 0; k < KTOP; ++k) {
    float best = -1e30f; int bi = 0;
    for (int i = tid; i < NA_TOTAL; i += 256) {
      float v = sc[i];
      if (v > best) { best = v; bi = i; }
    }
    sV[tid] = best; sI[tid] = bi;
    __syncthreads();
    for (int s = 128; s > 0; s >>= 1) {
      if (tid < s && sV[tid + s] > sV[tid]) { sV[tid] = sV[tid + s]; sI[tid] = sI[tid + s]; }
      __syncthreads();
    }
    if (tid == 0) {
      int idx = sI[0];
      topS[c * KTOP + k] = sV[0];
      topI[c * KTOP + k] = idx;
      topB[(c * KTOP + k) * 4 + 0] = boxes[idx * 4 + 0];
      topB[(c * KTOP + k) * 4 + 1] = boxes[idx * 4 + 1];
      topB[(c * KTOP + k) * 4 + 2] = boxes[idx * 4 + 2];
      topB[(c * KTOP + k) * 4 + 3] = boxes[idx * 4 + 3];
      sc[idx] = -1e30f;
    }
    __syncthreads();
  }
}

// ---------------- per-class padded NMS + final output packing ----------------
__global__ __launch_bounds__(256) void nms_write(const float* __restrict__ topS,
                                                 const float* __restrict__ topB,
                                                 float* __restrict__ out)
{
  __shared__ float bx[KTOP * 4];
  __shared__ float ss[KTOP];
  __shared__ int   keep[KTOP];
  int c = blockIdx.x, tid = threadIdx.x;
  for (int i = tid; i < KTOP; i += 256) {
    float s = topS[c * KTOP + i];
    ss[i] = s;
    keep[i] = (s > CLS_TH) ? 1 : 0;
    #pragma unroll
    for (int t = 0; t < 4; ++t) bx[i * 4 + t] = topB[(c * KTOP + i) * 4 + t];
  }
  __syncthreads();
  for (int i = 0; i < KTOP; ++i) {
    if (keep[i]) {
      float x1 = bx[i * 4 + 0], y1 = bx[i * 4 + 1];
      float x2 = bx[i * 4 + 2], y2 = bx[i * 4 + 3];
      float areaI = (x2 - x1) * (y2 - y1);
      for (int j = tid; j < KTOP; j += 256) {
        if (j > i && keep[j]) {
          float xx1 = fmaxf(x1, bx[j * 4 + 0]);
          float yy1 = fmaxf(y1, bx[j * 4 + 1]);
          float xx2 = fminf(x2, bx[j * 4 + 2]);
          float yy2 = fminf(y2, bx[j * 4 + 3]);
          float inter = fmaxf(xx2 - xx1, 0.f) * fmaxf(yy2 - yy1, 0.f);
          float areaJ = (bx[j * 4 + 2] - bx[j * 4 + 0]) * (bx[j * 4 + 3] - bx[j * 4 + 1]);
          float uni = fmaxf(areaI + areaJ - inter, 1e-8f);
          if (inter / uni > NMS_TH) keep[j] = 0;
        }
      }
    }
    __syncthreads();
  }
  float* outS   = out;
  float* outCid = out + NUM_CLASSES * KTOP;
  float* outB   = out + 2 * NUM_CLASSES * KTOP;
  float* outK   = out + 6 * NUM_CLASSES * KTOP;
  for (int i = tid; i < KTOP; i += 256) {
    int kp = keep[i];
    outS[c * KTOP + i]   = kp ? ss[i] : 0.0f;
    outCid[c * KTOP + i] = (float)c;
    #pragma unroll
    for (int t = 0; t < 4; ++t) outB[(c * KTOP + i) * 4 + t] = kp ? bx[i * 4 + t] : 0.0f;
    outK[c * KTOP + i] = kp ? 1.0f : 0.0f;
  }
}

// ---------------- host orchestration ----------------
extern "C" void kernel_launch(void* const* d_in, const int* in_sizes, int n_in,
                              void* d_out, int out_size, void* d_ws, size_t ws_size,
                              hipStream_t stream)
{
  (void)in_sizes; (void)n_in; (void)out_size; (void)ws_size;
  const float* p[5] = {(const float*)d_in[0], (const float*)d_in[1], (const float*)d_in[2],
                       (const float*)d_in[3], (const float*)d_in[4]};
  const float* reg_w  = (const float*)d_in[5];
  const float* reg_b  = (const float*)d_in[6];
  const float* reg_wo = (const float*)d_in[7];
  const float* reg_bo = (const float*)d_in[8];
  const float* cls_w  = (const float*)d_in[9];
  const float* cls_b  = (const float*)d_in[10];
  const float* cls_wo = (const float*)d_in[11];
  const float* cls_bo = (const float*)d_in[12];

  char* ws = (char*)d_ws;
  size_t off = 0;
  auto take = [&](size_t bytes) -> char* {
    char* pcur = ws + off;
    off = (off + bytes + 255) & ~(size_t)255;
    return pcur;
  };
  __bf16* wRegH  = (__bf16*)take(4ull * 256 * KDIM * 2);
  __bf16* wClsH  = (__bf16*)take(4ull * 256 * KDIM * 2);
  __bf16* wRegO  = (__bf16*)take(36ull * KDIM * 2);
  __bf16* wClsO  = (__bf16*)take(720ull * KDIM * 2);
  __bf16* actInB = (__bf16*)take(256ull * 4096 * 2);
  __bf16* actA   = (__bf16*)take(256ull * 4096 * 2);
  __bf16* actB   = (__bf16*)take(256ull * 4096 * 2);
  float*  regOut = (float*)take((size_t)NA_TOTAL * 4 * 4);
  float*  clsOut = (float*)take((size_t)NA_TOTAL * NUM_CLASSES * 4);
  float*  clsT   = (float*)take((size_t)NA_TOTAL * NUM_CLASSES * 4);
  float*  boxes  = (float*)take((size_t)NA_TOTAL * 4 * 4);
  float*  topS   = (float*)take((size_t)NUM_CLASSES * KTOP * 4);
  float*  topB   = (float*)take((size_t)NUM_CLASSES * KTOP * 4 * 4);
  int*    topI   = (int*)take((size_t)NUM_CLASSES * KTOP * 4);

  auto cvt = [&](const float* s, __bf16* d, size_t n) {
    f32_to_bf16<<<(unsigned)((n + 255) / 256), 256, 0, stream>>>(s, d, (int)n);
  };
  cvt(reg_w,  wRegH, 4ull * 256 * KDIM);
  cvt(cls_w,  wClsH, 4ull * 256 * KDIM);
  cvt(reg_wo, wRegO, 36ull * KDIM);
  cvt(cls_wo, wClsO, 720ull * KDIM);

  const int Hs[5]    = {64, 32, 16, 8, 4};
  const int aoffs[5] = {0, 36864, 46080, 48384, 48960};
  const size_t LYR = (size_t)256 * KDIM;

  for (int l = 0; l < 5; ++l) {
    int H = Hs[l], HW = H * H;
    cvt(p[l], actInB, (size_t)256 * HW);
    int nt = (HW + TILE_N - 1) / TILE_N;
    dim3 gh(nt, 2);        // 256 out channels
    dim3 gro(nt, 1);       // 36 out channels
    dim3 gco(nt, 6);       // 720 out channels (45 tiles)
    // regression head
    conv3x3_wmma<<<gh, 256, 0, stream>>>(actInB, wRegH + 0 * LYR, reg_b + 0 * 256, H, 256, 0, actA, nullptr, 0);
    conv3x3_wmma<<<gh, 256, 0, stream>>>(actA,   wRegH + 1 * LYR, reg_b + 1 * 256, H, 256, 0, actB, nullptr, 0);
    conv3x3_wmma<<<gh, 256, 0, stream>>>(actB,   wRegH + 2 * LYR, reg_b + 2 * 256, H, 256, 0, actA, nullptr, 0);
    conv3x3_wmma<<<gh, 256, 0, stream>>>(actA,   wRegH + 3 * LYR, reg_b + 3 * 256, H, 256, 0, actB, nullptr, 0);
    conv3x3_wmma<<<gro, 256, 0, stream>>>(actB,  wRegO, reg_bo, H, 36, 1, nullptr, regOut, aoffs[l]);
    // classification head
    conv3x3_wmma<<<gh, 256, 0, stream>>>(actInB, wClsH + 0 * LYR, cls_b + 0 * 256, H, 256, 0, actA, nullptr, 0);
    conv3x3_wmma<<<gh, 256, 0, stream>>>(actA,   wClsH + 1 * LYR, cls_b + 1 * 256, H, 256, 0, actB, nullptr, 0);
    conv3x3_wmma<<<gh, 256, 0, stream>>>(actB,   wClsH + 2 * LYR, cls_b + 2 * 256, H, 256, 0, actA, nullptr, 0);
    conv3x3_wmma<<<gh, 256, 0, stream>>>(actA,   wClsH + 3 * LYR, cls_b + 3 * 256, H, 256, 0, actB, nullptr, 0);
    conv3x3_wmma<<<gco, 256, 0, stream>>>(actB,  wClsO, cls_bo, H, 720, 2, nullptr, clsOut, aoffs[l]);
  }

  decode_boxes<<<(NA_TOTAL + 255) / 256, 256, 0, stream>>>(regOut, boxes);
  mask_transpose<<<(unsigned)(((size_t)NA_TOTAL * NUM_CLASSES + 255) / 256), 256, 0, stream>>>(clsOut, clsT, NA_TOTAL);
  topk_select<<<NUM_CLASSES, 256, 0, stream>>>(clsT, boxes, topS, topB, topI);
  nms_write<<<NUM_CLASSES, 256, 0, stream>>>(topS, topB, (float*)d_out);
}
